// Transformer_39676907882507
// MI455X (gfx1250) — matrix operand order repaired and checked by reading, not compiled, not run on hardware
//
#include <hip/hip_runtime.h>

#define BB 8
#define LQn 2048
#define LKn 2048
#define DDn 512
#define DVn 512

typedef __attribute__((ext_vector_type(16))) __bf16 v16bf;
typedef __attribute__((ext_vector_type(8)))  float  v8f;

union Frag16 { uint4 q[2]; unsigned u[8]; v16bf v; };

__device__ __forceinline__ unsigned short f2bf(float f) {
  unsigned u = __float_as_uint(f);
  u += 0x7FFFu + ((u >> 16) & 1u);          // round-to-nearest-even
  return (unsigned short)(u >> 16);
}
__device__ __forceinline__ unsigned pack2(float a, float b) {
  return (unsigned)f2bf(a) | ((unsigned)f2bf(b) << 16);
}

// ---------------- Kernel A: row sums over D=512 (one wave per row) ----------
__global__ __launch_bounds__(256) void rowsum512(const float* __restrict__ src,
                                                 float* __restrict__ dst, int nrows) {
  int wave = blockIdx.x * (blockDim.x >> 5) + (threadIdx.x >> 5);
  int lane = threadIdx.x & 31;
  if (wave >= nrows) return;
  const float4* p = (const float4*)(src + (size_t)wave * DDn);
  float s = 0.f;
#pragma unroll
  for (int i = 0; i < 4; ++i) {
    float4 v = p[i * 32 + lane];
    s += v.x + v.y + v.z + v.w;
  }
#pragma unroll
  for (int off = 16; off; off >>= 1) s += __shfl_down(s, off, 32);
  if (lane == 0) dst[wave] = s;
}

// ---------------- Kernel B1: f32 -> bf16 row-major (K) ----------------------
__global__ __launch_bounds__(256) void cvt_bf16(const float* __restrict__ src,
                                                unsigned short* __restrict__ dst, int n4) {
  int i = blockIdx.x * blockDim.x + threadIdx.x;
  if (i >= n4) return;
  float4 v = ((const float4*)src)[i];
  ((uint2*)dst)[i] = make_uint2(pack2(v.x, v.y), pack2(v.z, v.w));
}

// ---------------- Kernel B2: V[b][k][n] -> bf16 Vt[b][n][k] -----------------
__global__ __launch_bounds__(256) void vt_bf16(const float* __restrict__ V,
                                               unsigned short* __restrict__ Vt, int total) {
  int i = blockIdx.x * blockDim.x + threadIdx.x;
  if (i >= total) return;                 // i = (b*DV + n)*(LK/2) + k2
  int k2 = i & (LKn / 2 - 1);
  int t  = i >> 10;                       // / (LK/2)
  int n  = t & (DVn - 1);
  int b  = t >> 9;                        // / DV
  size_t base = ((size_t)b * LKn + (size_t)(2 * k2)) * DVn + n;
  ((unsigned*)Vt)[i] = pack2(V[base], V[base + DVn]);
}

// ---------------- Kernel C: per-row softmax max & denominator ---------------
__global__ __launch_bounds__(256) void softstats(const float* __restrict__ mask,
                                                 const float* __restrict__ qs,
                                                 const float* __restrict__ ks,
                                                 float* __restrict__ M, float* __restrict__ L) {
  int row = blockIdx.x;                   // b*LQ + q
  int b = row >> 11;
  int t = threadIdx.x;
  const float scale = 0.044194173824159216f; // 1/sqrt(512)
  float qv = qs[row];
  const float* mrow = mask + (size_t)row * LKn;
  const float* krow = ks + b * LKn;
  float m = -3.0e38f, l = 0.f;
  for (int k = t; k < LKn; k += 256) {
    float s = (qv + krow[k]) * scale + mrow[k];
    if (s > m) { l = l * __expf(m - s) + 1.f; m = s; }
    else        l += __expf(s - m);
  }
  __shared__ float sm[256], sl[256];
  sm[t] = m; sl[t] = l;
  __syncthreads();
  for (int str = 128; str; str >>= 1) {
    if (t < str) {
      float m2 = sm[t + str], l2 = sl[t + str];
      float mn = fmaxf(sm[t], m2);
      sl[t] = sl[t] * __expf(sm[t] - mn) + l2 * __expf(m2 - mn);
      sm[t] = mn;
    }
    __syncthreads();
  }
  if (t == 0) { M[row] = sm[0]; L[row] = sl[0]; }
}

// ---------------- Kernel D: main  (S = Q*K^T, P = w.*S, O = P*V) ------------
#define QRS (DDn + 8)     // padded Q_lds row stride (bf16 elems), 16B-aligned rows
#define PRS (128 + 8)     // padded P_lds row stride

__global__ __launch_bounds__(256) void attn_main(
    const float* __restrict__ Q, const float* __restrict__ mask,
    const unsigned short* __restrict__ Kbf, const unsigned short* __restrict__ Vt,
    const float* __restrict__ qs, const float* __restrict__ ks,
    const float* __restrict__ Mr, const float* __restrict__ Lr,
    float* __restrict__ out) {
  __shared__ __align__(16) unsigned short Qs[16 * QRS];
  __shared__ __align__(16) unsigned short Ps[16 * PRS];

  int blk = blockIdx.x;           // b*(LQ/16) + qt
  int b  = blk >> 7;
  int q0 = (blk & 127) << 4;
  int tid = threadIdx.x;
  int w = tid >> 5, l = tid & 31;
  int h = l >> 4, nn = l & 15;

  // ---- stage Q tile (16 x 512) into LDS as bf16, padded rows ----
  {
    const float2* qsrc = (const float2*)(Q + (size_t)(b * LQn + q0) * DDn);
    for (int i = tid; i < 16 * 256; i += 256) {
      int r = i >> 8, c = i & 255;
      float2 v = qsrc[r * 256 + c];
      *(unsigned*)&Qs[r * QRS + 2 * c] = pack2(v.x, v.y);
    }
  }
  // per-row softmax stats for the 8 C-rows this lane owns (M = 8*h + j)
  float qsv[8], Mv[8], invL[8];
#pragma unroll
  for (int j = 0; j < 8; ++j) {
    int row = b * LQn + q0 + 8 * h + j;
    qsv[j] = qs[row]; Mv[j] = Mr[row]; invL[j] = 1.f / Lr[row];
  }
  const float scale = 0.044194173824159216f;

  v8f oacc[4] = {};
  __syncthreads();

  const unsigned short* kbase = Kbf + (size_t)b * LKn * DDn;
  const unsigned short* vbase = Vt + (size_t)b * DVn * LKn;

  for (int kt = 0; kt < LKn / 128; ++kt) {
    int key0 = kt * 128;
    int n0 = key0 + w * 16;                       // this wave's 16 keys
    float ksv = ks[b * LKn + n0 + nn];
    const unsigned short* krow = kbase + (size_t)(n0 + nn) * DDn;
    if (kt + 1 < LKn / 128)                       // pull next K tile toward L2/L0
      __builtin_prefetch(kbase + (size_t)(n0 + 128 + nn) * DDn, 0, 1);

    // ---- S = Q * K^T  (16 bf16 WMMAs over D=512) ----
    v8f c = {};
#pragma unroll
    for (int d0 = 0; d0 < DDn; d0 += 32) {
      Frag16 a, bf;
      const unsigned short* qrow = Qs + nn * QRS;         // A: lane row = l&15
      a.q[0]  = *(const uint4*)(qrow + d0 + h * 8);       // K = h*8 + [0..7]
      a.q[1]  = *(const uint4*)(qrow + d0 + 16 + h * 8);  // K = 16 + h*8 + [0..7]
      bf.q[0] = *(const uint4*)(krow + d0 + h * 8);       // B: lane col = l&15
      bf.q[1] = *(const uint4*)(krow + d0 + 16 + h * 8);  // K = h*8 + [0..7], 16 + h*8 + [0..7]
      c = __builtin_amdgcn_wmma_f32_16x16x32_bf16(false, a.v, false, bf.v,
                                                  (short)0, c, false, false);
    }

    // ---- P = softmax_weight .* S  ->  LDS (bf16) ----
    const float* mrow0 = mask + (size_t)(b * LQn + q0) * LKn + n0 + nn;
#pragma unroll
    for (int j = 0; j < 8; ++j) {
      int m = 8 * h + j;                                   // C-row for element j
      float s = (qsv[j] + ksv) * scale + mrow0[m * LKn];
      float wgt = __expf(s - Mv[j]) * invL[j];
      Ps[m * PRS + w * 16 + nn] = f2bf(wgt * c[j]);
    }
    __syncthreads();

    // ---- O += P * V   (this wave owns DV columns [w*64, w*64+64) ) ----
#pragma unroll
    for (int kk = 0; kk < 4; ++kk) {
      Frag16 a;
      const unsigned short* prow = Ps + nn * PRS;
      int d0 = kk * 32;
      a.q[0] = *(const uint4*)(prow + d0 + h * 8);
      a.q[1] = *(const uint4*)(prow + d0 + 16 + h * 8);
#pragma unroll
      for (int nt = 0; nt < 4; ++nt) {
        int col = w * 64 + nt * 16 + nn;
        const unsigned short* vrow =
            vbase + (size_t)col * LKn + key0 + kk * 32 + h * 8;
        Frag16 bv;
        bv.q[0] = *(const uint4*)(vrow);
        bv.q[1] = *(const uint4*)(vrow + 16);
        oacc[nt] = __builtin_amdgcn_wmma_f32_16x16x32_bf16(false, a.v, false, bv.v,
                                                           (short)0, oacc[nt], false, false);
      }
    }
    __syncthreads();
  }

  // ---- epilogue: C layout -> global f32 ----
#pragma unroll
  for (int nt = 0; nt < 4; ++nt) {
#pragma unroll
    for (int j = 0; j < 8; ++j) {
      int row = q0 + 8 * h + j;
      int col = w * 64 + nt * 16 + nn;
      out[(size_t)(b * LQn + row) * DVn + col] = oacc[nt][j];
    }
  }
}

// ---------------------------------------------------------------------------
extern "C" void kernel_launch(void* const* d_in, const int* in_sizes, int n_in,
                              void* d_out, int out_size, void* d_ws, size_t ws_size,
                              hipStream_t stream) {
  const float* Q    = (const float*)d_in[0];
  const float* K    = (const float*)d_in[1];
  const float* V    = (const float*)d_in[2];
  const float* mask = (const float*)d_in[3];
  float* out = (float*)d_out;

  char* ws = (char*)d_ws;
  unsigned short* Kbf = (unsigned short*)(ws);                      // 16 MB
  unsigned short* Vt  = (unsigned short*)(ws + (size_t)16777216);   // 16 MB
  float* qs = (float*)(ws + (size_t)33554432);                      // 64 KB
  float* ks = (float*)(ws + (size_t)33619968);                      // 64 KB
  float* Mr = (float*)(ws + (size_t)33685504);                      // 64 KB
  float* Lr = (float*)(ws + (size_t)33751040);                      // 64 KB

  rowsum512<<<(BB * LQn) / 8, 256, 0, stream>>>(Q, qs, BB * LQn);
  rowsum512<<<(BB * LKn) / 8, 256, 0, stream>>>(K, ks, BB * LKn);
  cvt_bf16<<<(BB * LKn * DDn / 4) / 256, 256, 0, stream>>>(K, Kbf, BB * LKn * DDn / 4);
  vt_bf16<<<(BB * DVn * (LKn / 2)) / 256, 256, 0, stream>>>(V, Vt, BB * DVn * (LKn / 2));
  softstats<<<BB * LQn, 256, 0, stream>>>(mask, qs, ks, Mr, Lr);
  attn_main<<<BB * (LQn / 16), 256, 0, stream>>>(Q, mask, Kbf, Vt, qs, ks, Mr, Lr, out);
}
